// ShowAttendAndTell_52226802320075
// MI455X (gfx1250) — matrix operand order repaired and checked by reading, not compiled, not run on hardware
//

#include <hip/hip_runtime.h>
#include <hip/hip_bf16.h>
#include <math.h>

// Model dims (fixed by reference)
#define Bn   64
#define Tn   32
#define TS   31      // T-1 steps
#define EMB  300
#define FEAT 128
#define HID  512
#define Pn   256
#define Vn   30000

typedef __attribute__((ext_vector_type(16))) _Float16 v16h_t;
typedef __attribute__((ext_vector_type(8)))  _Float16 v8h_t;
typedef __attribute__((ext_vector_type(8)))  float    v8f_t;

union V16U { v16h_t v; v8h_t h[2]; };

// ---------------- utility kernels ----------------

__global__ void k_zero_f32(float* __restrict__ p, int n) {
    int i = blockIdx.x * blockDim.x + threadIdx.x;
    if (i < n) p[i] = 0.0f;
}

__global__ void k_f32_to_f16(const float* __restrict__ src, _Float16* __restrict__ dst, int n) {
    int i = blockIdx.x * blockDim.x + threadIdx.x;
    if (i < n) dst[i] = (_Float16)src[i];
}

// td_target[b,i] = sum_f target_feat[b,f] * W_td2[i,f]   (B x 128, dot 128)
__global__ void k_td_target(const float* __restrict__ tf, const float* __restrict__ W_td2,
                            float* __restrict__ out) {
    int idx = blockIdx.x * blockDim.x + threadIdx.x;   // b*128 + i
    int b = idx >> 7, i = idx & 127;
    const float* t = tf + b * FEAT;
    const float* w = W_td2 + i * FEAT;
    float acc = 0.f;
    #pragma unroll 4
    for (int k = 0; k < FEAT; ++k) acc += t[k] * w[k];
    out[idx] = acc;
}

// obj_proj[b,p,h] = sum_f obj_feats[b,p,f] * W_feat[h,f]
__global__ void k_obj_proj(const float* __restrict__ of, const float* __restrict__ W_feat,
                           float* __restrict__ out) {
    size_t idx = (size_t)blockIdx.x * blockDim.x + threadIdx.x;  // ((b*P)+p)*H + h
    int h = (int)(idx & (HID - 1));
    size_t bp = idx >> 9;
    const float* o = of + bp * FEAT;
    const float* w = W_feat + h * FEAT;
    float acc = 0.f;
    #pragma unroll 4
    for (int k = 0; k < FEAT; ++k) acc += o[k] * w[k];
    out[idx] = acc;
}

// ---------------- per-step fused small kernels ----------------

// s = relu( tanh(emb_t @ W_td3.T + h2 @ W_td1.T + td_target) @ W_td.T )
// grid: B blocks x 128 threads
__global__ void k_s(const float* __restrict__ embs, int t,
                    const float* __restrict__ h2,
                    const float* __restrict__ W_td3, const float* __restrict__ W_td1,
                    const float* __restrict__ tdt, const float* __restrict__ W_td,
                    float* __restrict__ s) {
    int b = blockIdx.x, i = threadIdx.x;
    __shared__ float th[128];
    const float* e  = embs + ((size_t)b * Tn + t) * EMB;
    const float* w3 = W_td3 + i * EMB;
    float acc = tdt[b * 128 + i];
    #pragma unroll 4
    for (int k = 0; k < EMB; ++k) acc += e[k] * w3[k];
    const float* hb = h2 + b * HID;
    const float* w1 = W_td1 + i * HID;
    #pragma unroll 4
    for (int k = 0; k < HID; ++k) acc += hb[k] * w1[k];
    th[i] = tanhf(acc);
    __syncthreads();
    const float* wt = W_td + i * 128;
    float a2 = 0.f;
    #pragma unroll 4
    for (int k = 0; k < 128; ++k) a2 += th[k] * wt[k];
    s[b * 128 + i] = fmaxf(a2, 0.f);
}

// PyTorch GRUCell. One thread per (b,j). xdim = 128 for both GRUs.
// Writes f32 hout (and optional f16 copy for the WMMA classifier).
__global__ void k_gru(const float* __restrict__ x, int xdim,
                      const float* __restrict__ h,
                      const float* __restrict__ Wih, const float* __restrict__ Whh,
                      const float* __restrict__ bih, const float* __restrict__ bhh,
                      float* __restrict__ hout, _Float16* __restrict__ houth) {
    int idx = blockIdx.x * blockDim.x + threadIdx.x;   // b*HID + j
    int b = idx >> 9, j = idx & (HID - 1);
    const float* xb = x + b * xdim;
    const float* hb = h + b * HID;
    float gi[3], gh[3];
    #pragma unroll
    for (int g = 0; g < 3; ++g) {
        int row = g * HID + j;
        const float* wi = Wih + (size_t)row * xdim;
        float a = bih[row];
        #pragma unroll 4
        for (int k = 0; k < 128; ++k) a += xb[k] * wi[k];
        gi[g] = a;
        const float* wh = Whh + (size_t)row * HID;
        float c = bhh[row];
        #pragma unroll 4
        for (int k = 0; k < HID; ++k) c += hb[k] * wh[k];
        gh[g] = c;
    }
    float r = 1.f / (1.f + __expf(-(gi[0] + gh[0])));
    float z = 1.f / (1.f + __expf(-(gi[1] + gh[1])));
    float n = tanhf(gi[2] + r * gh[2]);
    float hv = (1.f - z) * n + z * hb[j];
    hout[idx] = hv;
    if (houth) houth[idx] = (_Float16)hv;
}

// h1w[b,h] = sum_k h1[b,k] * W_hidd[h,k]
__global__ void k_h1w(const float* __restrict__ h1, const float* __restrict__ W_hidd,
                      float* __restrict__ out) {
    int idx = blockIdx.x * blockDim.x + threadIdx.x;   // b*HID + h
    int b = idx >> 9, hh = idx & (HID - 1);
    const float* hb = h1 + b * HID;
    const float* w = W_hidd + (size_t)hh * HID;
    float acc = 0.f;
    #pragma unroll 4
    for (int k = 0; k < HID; ++k) acc += hb[k] * w[k];
    out[idx] = acc;
}

// attention: scores -> softmax -> mask out + attended.  grid: B blocks x 256 threads
__global__ void k_attn(const float* __restrict__ obj_proj, const float* __restrict__ h1w,
                       const float* __restrict__ W_attend, const float* __restrict__ obj_feats,
                       float* __restrict__ attended, float* __restrict__ topdown, int t) {
    int b = blockIdx.x, p = threadIdx.x;
    __shared__ float sc[Pn];
    __shared__ float red[Pn];
    const float* op = obj_proj + ((size_t)b * Pn + p) * HID;
    const float* hw = h1w + b * HID;
    float sco = 0.f;
    #pragma unroll 4
    for (int h = 0; h < HID; ++h) sco += tanhf(op[h] + hw[h]) * W_attend[h];
    sc[p] = sco; red[p] = sco;
    __syncthreads();
    for (int off = 128; off > 0; off >>= 1) {
        if (p < off) red[p] = fmaxf(red[p], red[p + off]);
        __syncthreads();
    }
    float mx = red[0];
    __syncthreads();
    float e = __expf(sco - mx);
    sc[p] = e; red[p] = e;
    __syncthreads();
    for (int off = 128; off > 0; off >>= 1) {
        if (p < off) red[p] += red[p + off];
        __syncthreads();
    }
    float m = e / red[0];
    sc[p] = m;
    topdown[((size_t)b * Pn + p) * TS + t] = m;
    __syncthreads();
    if (p < FEAT) {
        const float* of = obj_feats + (size_t)b * Pn * FEAT + p;
        float a = 0.f;
        #pragma unroll 4
        for (int q = 0; q < Pn; ++q) a += of[(size_t)q * FEAT] * sc[q];
        attended[b * FEAT + p] = a;
    }
}

// l = relu( tanh(attended @ W_l1.T + h1 @ W_l2.T) @ W_l.T )   grid: B x 128
__global__ void k_l(const float* __restrict__ att, const float* __restrict__ h1,
                    const float* __restrict__ W_l1, const float* __restrict__ W_l2,
                    const float* __restrict__ W_l, float* __restrict__ l) {
    int b = blockIdx.x, i = threadIdx.x;
    __shared__ float th[128];
    const float* ab = att + b * FEAT;
    const float* w1 = W_l1 + i * FEAT;
    float acc = 0.f;
    #pragma unroll 4
    for (int k = 0; k < FEAT; ++k) acc += ab[k] * w1[k];
    const float* hb = h1 + b * HID;
    const float* w2 = W_l2 + i * HID;
    #pragma unroll 4
    for (int k = 0; k < HID; ++k) acc += hb[k] * w2[k];
    th[i] = tanhf(acc);
    __syncthreads();
    const float* wl = W_l + i * 128;
    float a2 = 0.f;
    #pragma unroll 4
    for (int k = 0; k < 128; ++k) a2 += th[k] * wl[k];
    l[b * 128 + i] = fmaxf(a2, 0.f);
}

// ---------------- WMMA classifier: logits = h2 @ W_cls.T + b_cls ----------------
// M=64(B) x N=30000(V) x K=512, f16 inputs, f32 accum.
// Register-blocked: each wave computes a 32(M) x 48(N) output block:
//   2 A fragments x 3 B fragments -> 6 v_wmma_f32_16x16x32_f16 per K-step,
//   10 b128 loads per 6 WMMAs (vs 4 loads per WMMA unblocked) => ~2.4x less
//   L2 traffic per FLOP on the compute-dominant GEMM (W_cls f16 is L2-resident).
// Tiling exact: M=64 -> 2 row-blocks of 32; N=30000 = 625*48; K=512 = 16*32.
// Fragment layouts used here (contiguous K, same order for A and B):
//   A (16x32 f16): lane L -> row M=L%16; lanes 0-15 K 0..15, lanes 16-31 K 16..31
//   B (32x16 f16): lane L -> col N=L%16; lanes 0-15 K 0..15, lanes 16-31 K 16..31
__global__ void k_cls_wmma(const _Float16* __restrict__ h2h,   // B x 512
                           const _Float16* __restrict__ Wh,    // V x 512
                           const float* __restrict__ bias,     // V
                           float* __restrict__ lang_cap,       // B x TS x V
                           int t) {
    int lane = threadIdx.x & 31;
    int w = blockIdx.x * 2 + (threadIdx.x >> 5);   // 1250 waves total
    int mhalf = w & 1;                              // 0 or 1 -> m0 = 0 or 32
    int ngrp  = w >> 1;                             // 0..624  -> n0 = ngrp*48
    int m0 = mhalf * 32;
    int n0 = ngrp * 48;
    int lrow   = lane & 15;
    int hihalf = lane >> 4;
    int akb = hihalf * 16;    // A K-sub-offset per half-wave
    int bkb = hihalf * 16;    // B K-sub-offset per half-wave

    const _Float16* arow0 = h2h + (size_t)(m0 + lrow) * HID;
    const _Float16* arow1 = h2h + (size_t)(m0 + 16 + lrow) * HID;
    const _Float16* brow0 = Wh + (size_t)(n0 + lrow) * HID + bkb;
    const _Float16* brow1 = Wh + (size_t)(n0 + 16 + lrow) * HID + bkb;
    const _Float16* brow2 = Wh + (size_t)(n0 + 32 + lrow) * HID + bkb;

    v8f_t acc00 = {}, acc01 = {}, acc02 = {};
    v8f_t acc10 = {}, acc11 = {}, acc12 = {};

    for (int k = 0; k < HID; k += 32) {
        V16U a0, a1;
        a0.h[0] = *(const v8h_t*)(arow0 + k + akb);
        a0.h[1] = *(const v8h_t*)(arow0 + k + 8 + akb);
        a1.h[0] = *(const v8h_t*)(arow1 + k + akb);
        a1.h[1] = *(const v8h_t*)(arow1 + k + 8 + akb);
        v16h_t b0 = *(const v16h_t*)(brow0 + k);
        v16h_t b1 = *(const v16h_t*)(brow1 + k);
        v16h_t b2 = *(const v16h_t*)(brow2 + k);
        if (k + 64 <= HID - 32) {
            __builtin_prefetch(brow0 + k + 64, 0, 0);
            __builtin_prefetch(brow1 + k + 64, 0, 0);
            __builtin_prefetch(brow2 + k + 64, 0, 0);
        }
        acc00 = __builtin_amdgcn_wmma_f32_16x16x32_f16(false, a0.v, false, b0, (short)0, acc00, false, false);
        acc01 = __builtin_amdgcn_wmma_f32_16x16x32_f16(false, a0.v, false, b1, (short)0, acc01, false, false);
        acc02 = __builtin_amdgcn_wmma_f32_16x16x32_f16(false, a0.v, false, b2, (short)0, acc02, false, false);
        acc10 = __builtin_amdgcn_wmma_f32_16x16x32_f16(false, a1.v, false, b0, (short)0, acc10, false, false);
        acc11 = __builtin_amdgcn_wmma_f32_16x16x32_f16(false, a1.v, false, b1, (short)0, acc11, false, false);
        acc12 = __builtin_amdgcn_wmma_f32_16x16x32_f16(false, a1.v, false, b2, (short)0, acc12, false, false);
    }

    // C layout: VGPR r, lane L: M = r + 8*(L>>4), N = L%16
    int ncol = lane & 15;
    int mhi  = hihalf * 8;
    float bv0 = bias[n0 + ncol];
    float bv1 = bias[n0 + 16 + ncol];
    float bv2 = bias[n0 + 32 + ncol];
    #pragma unroll
    for (int r = 0; r < 8; ++r) {
        int bA = m0 + mhi + r;          // rows m0..m0+15
        int bB = m0 + 16 + mhi + r;     // rows m0+16..m0+31
        float* oA = lang_cap + ((size_t)bA * TS + t) * Vn + n0;
        float* oB = lang_cap + ((size_t)bB * TS + t) * Vn + n0;
        oA[ncol]      = acc00[r] + bv0;
        oA[16 + ncol] = acc01[r] + bv1;
        oA[32 + ncol] = acc02[r] + bv2;
        oB[ncol]      = acc10[r] + bv0;
        oB[16 + ncol] = acc11[r] + bv1;
        oB[32 + ncol] = acc12[r] + bv2;
    }
}

// ---------------- host-side orchestration ----------------

extern "C" void kernel_launch(void* const* d_in, const int* in_sizes, int n_in,
                              void* d_out, int out_size, void* d_ws, size_t ws_size,
                              hipStream_t stream) {
    const float* word_embs   = (const float*)d_in[0];
    const float* target_feat = (const float*)d_in[1];
    const float* obj_feats   = (const float*)d_in[2];
    const float* W_td1       = (const float*)d_in[3];
    const float* W_td2       = (const float*)d_in[4];
    const float* W_td3       = (const float*)d_in[5];
    const float* W_td        = (const float*)d_in[6];
    const float* g1_Wih      = (const float*)d_in[7];
    const float* g1_Whh      = (const float*)d_in[8];
    const float* g1_bih      = (const float*)d_in[9];
    const float* g1_bhh      = (const float*)d_in[10];
    const float* W_feat      = (const float*)d_in[11];
    const float* W_hidd      = (const float*)d_in[12];
    const float* W_attend    = (const float*)d_in[13];
    const float* W_l1        = (const float*)d_in[14];
    const float* W_l2        = (const float*)d_in[15];
    const float* W_l         = (const float*)d_in[16];
    const float* g2_Wih      = (const float*)d_in[17];
    const float* g2_Whh      = (const float*)d_in[18];
    const float* g2_bih      = (const float*)d_in[19];
    const float* g2_bhh      = (const float*)d_in[20];
    const float* W_cls       = (const float*)d_in[21];
    const float* b_cls       = (const float*)d_in[22];

    float* lang_cap = (float*)d_out;                                // B x TS x V
    float* topdown  = lang_cap + (size_t)Bn * TS * Vn;              // B x P x TS

    // workspace carve-up (~66 MB)
    char* ws = (char*)d_ws;
    size_t off = 0;
    auto alloc = [&](size_t bytes) { char* p = ws + off; off += (bytes + 255) & ~(size_t)255; return p; };
    _Float16* Wh       = (_Float16*)alloc((size_t)Vn * HID * 2);   // 30.7 MB, L2-resident
    float*    obj_proj = (float*)   alloc((size_t)Bn * Pn * HID * 4); // 33.5 MB
    float*    tdt      = (float*)   alloc((size_t)Bn * 128 * 4);
    float*    h1a      = (float*)   alloc((size_t)Bn * HID * 4);
    float*    h1b      = (float*)   alloc((size_t)Bn * HID * 4);
    float*    h2a      = (float*)   alloc((size_t)Bn * HID * 4);
    float*    h2b      = (float*)   alloc((size_t)Bn * HID * 4);
    _Float16* h2h      = (_Float16*)alloc((size_t)Bn * HID * 2);
    float*    sbuf     = (float*)   alloc((size_t)Bn * 128 * 4);
    float*    lbuf     = (float*)   alloc((size_t)Bn * 128 * 4);
    float*    h1w      = (float*)   alloc((size_t)Bn * HID * 4);
    float*    attended = (float*)   alloc((size_t)Bn * FEAT * 4);
    (void)ws_size; (void)in_sizes; (void)n_in; (void)out_size;

    // one-time precompute
    {
        int n = Vn * HID;
        k_f32_to_f16<<<(n + 255) / 256, 256, 0, stream>>>(W_cls, Wh, n);
        k_zero_f32<<<(Bn * HID + 255) / 256, 256, 0, stream>>>(h1a, Bn * HID);
        k_zero_f32<<<(Bn * HID + 255) / 256, 256, 0, stream>>>(h2a, Bn * HID);
        k_td_target<<<(Bn * 128) / 128, 128, 0, stream>>>(target_feat, W_td2, tdt);
        size_t n2 = (size_t)Bn * Pn * HID;
        k_obj_proj<<<(unsigned)(n2 / 256), 256, 0, stream>>>(obj_feats, W_feat, obj_proj);
    }

    float* h1c = h1a; float* h1n = h1b;
    float* h2c = h2a; float* h2n = h2b;

    for (int t = 0; t < TS; ++t) {
        k_s<<<Bn, 128, 0, stream>>>(word_embs, t, h2c, W_td3, W_td1, tdt, W_td, sbuf);
        k_gru<<<(Bn * HID) / 256, 256, 0, stream>>>(sbuf, 128, h1c, g1_Wih, g1_Whh,
                                                    g1_bih, g1_bhh, h1n, (_Float16*)nullptr);
        k_h1w<<<(Bn * HID) / 256, 256, 0, stream>>>(h1n, W_hidd, h1w);
        k_attn<<<Bn, Pn, 0, stream>>>(obj_proj, h1w, W_attend, obj_feats, attended, topdown, t);
        k_l<<<Bn, 128, 0, stream>>>(attended, h1n, W_l1, W_l2, W_l, lbuf);
        k_gru<<<(Bn * HID) / 256, 256, 0, stream>>>(lbuf, 128, h2c, g2_Wih, g2_Whh,
                                                    g2_bih, g2_bhh, h2n, h2h);
        // 1250 tile-waves (2 per 64-thread block), each computing a 32x48 block
        k_cls_wmma<<<625, 64, 0, stream>>>(h2h, Wh, b_cls, lang_cap, t);

        float* tmp;
        tmp = h1c; h1c = h1n; h1n = tmp;
        tmp = h2c; h2c = h2n; h2n = tmp;
    }
}
